// MultiHeadedAttention_81441169866856
// MI455X (gfx1250) — matrix operand order repaired and checked by reading, not compiled, not run on hardware
//
#include <hip/hip_runtime.h>

typedef __attribute__((ext_vector_type(16))) __bf16 v16bf;
typedef __attribute__((ext_vector_type(8)))  __bf16 v8bf;
typedef __attribute__((ext_vector_type(8)))  float  v8f;
typedef __attribute__((ext_vector_type(4)))  int    v4i;

#define NUM_HEADS 16
#define HEAD_SIZE 64
#define SEQ_LEN   2048
#define D_MODEL   1024
#define BATCH     2
#define ROWS      (BATCH * SEQ_LEN)          /* 4096 */
#define WELEMS    (D_MODEL * D_MODEL)        /* 1M   */

// ---------- helpers ----------

__device__ __forceinline__ unsigned short f2bf(float f) {
  unsigned int u = __builtin_bit_cast(unsigned int, f);
  u += 0x7FFFu + ((u >> 16) & 1u);          // round-to-nearest-even
  return (unsigned short)(u >> 16);
}

__device__ __forceinline__ v8f wmma_bf16(v16bf a, v16bf b, v8f c) {
  // D = A(16x32 bf16) * B(32x16 bf16) + C(16x16 f32)
  return __builtin_amdgcn_wmma_f32_16x16x32_bf16(false, a, false, b,
                                                 (short)0, c, false, false);
}

// A-operand gather: 16-bit A 16x32 layout -> per lane elements {base+0..15} (contiguous k)
__device__ __forceinline__ v16bf load_a32(const unsigned short* base) {
  v8bf lo = *reinterpret_cast<const v8bf*>(base);
  v8bf hi = *reinterpret_cast<const v8bf*>(base + 8);
  v16bf r;
#pragma unroll
  for (int i = 0; i < 8; ++i) { r[i] = lo[i]; r[i + 8] = hi[i]; }
  return r;
}

// ---------- async global->LDS staging (CDNA5 path, guarded) ----------

#if defined(__HIP_DEVICE_COMPILE__) && \
    __has_builtin(__builtin_amdgcn_global_load_async_to_lds_b128)
#define MHA_ASYNC_LDS 1
typedef __attribute__((address_space(3))) v4i* lds_v4i_p;
__device__ __forceinline__ void stage16(const unsigned short* g, unsigned short* l) {
  __builtin_amdgcn_global_load_async_to_lds_b128(
      (v4i*)g, (lds_v4i_p)l, 0, 0);
}
__device__ __forceinline__ void stage_fence() {
  asm volatile("s_wait_asynccnt 0" ::: "memory");
}
#else
__device__ __forceinline__ void stage16(const unsigned short* g, unsigned short* l) {
  *reinterpret_cast<v8bf*>(l) = *reinterpret_cast<const v8bf*>(g);
}
__device__ __forceinline__ void stage_fence() {}
#endif

// ---------- kernel 1: f32 -> bf16 conversion (+ weight transpose) ----------

__global__ void mha_convert_kernel(const float* __restrict__ x,
                                   const float* __restrict__ wq,
                                   const float* __restrict__ wk,
                                   const float* __restrict__ wv,
                                   const float* __restrict__ wo,
                                   unsigned short* __restrict__ xbf,
                                   unsigned short* __restrict__ wT) {
  const int i = blockIdx.x * blockDim.x + threadIdx.x;
  if (i < ROWS * D_MODEL) xbf[i] = f2bf(x[i]);
  if (i < WELEMS) {
    const int kk = i >> 10;           // input-feature index
    const int nn = i & 1023;          // output-column index
    const size_t dst = (size_t)nn * D_MODEL + kk;   // W^T[n][k]
    wT[0 * (size_t)WELEMS + dst] = f2bf(wq[i]);
    wT[1 * (size_t)WELEMS + dst] = f2bf(wk[i]);
    wT[2 * (size_t)WELEMS + dst] = f2bf(wv[i]);
    wT[3 * (size_t)WELEMS + dst] = f2bf(wo[i]);
  }
}

// ---------- GEMM core: one wave -> 32x64 tile, double-buffered K loop ----------
// acc[i][t]: rows rowbase+i*16+{0..15}, cols colbase+t*16+{0..15}

struct TileAcc { v8f a[2][4]; };

__device__ __forceinline__ TileAcc gemm_32x64(const unsigned short* __restrict__ abase,
                                              const unsigned short* __restrict__ wbase,
                                              int rowbase, int colbase,
                                              int half, int l16) {
  const unsigned short* ar0 = abase + (size_t)(rowbase + l16) * D_MODEL + half * 16;
  const unsigned short* ar1 = ar0 + (size_t)16 * D_MODEL;
  const unsigned short* bp0 = wbase + (size_t)(colbase + 0  + l16) * D_MODEL + half * 16;
  const unsigned short* bp1 = wbase + (size_t)(colbase + 16 + l16) * D_MODEL + half * 16;
  const unsigned short* bp2 = wbase + (size_t)(colbase + 32 + l16) * D_MODEL + half * 16;
  const unsigned short* bp3 = wbase + (size_t)(colbase + 48 + l16) * D_MODEL + half * 16;

  TileAcc acc;
#pragma unroll
  for (int i = 0; i < 2; ++i)
#pragma unroll
    for (int t = 0; t < 4; ++t) acc.a[i][t] = v8f{};

  // prologue: tiles for kt = 0
  v16bf a0 = load_a32(ar0), a1 = load_a32(ar1);
  v16bf b0 = *reinterpret_cast<const v16bf*>(bp0);
  v16bf b1 = *reinterpret_cast<const v16bf*>(bp1);
  v16bf b2 = *reinterpret_cast<const v16bf*>(bp2);
  v16bf b3 = *reinterpret_cast<const v16bf*>(bp3);

  for (int kt = 0; kt < D_MODEL; kt += 32) {
    const int nk = (kt + 32) & (D_MODEL - 1);   // wraps on final iter (dead value)
    __builtin_prefetch(ar0 + nk + 32, 0, 1);
    // issue next-iteration loads before consuming current tiles
    v16bf na0 = load_a32(ar0 + nk);
    v16bf na1 = load_a32(ar1 + nk);
    v16bf nb0 = *reinterpret_cast<const v16bf*>(bp0 + nk);
    v16bf nb1 = *reinterpret_cast<const v16bf*>(bp1 + nk);
    v16bf nb2 = *reinterpret_cast<const v16bf*>(bp2 + nk);
    v16bf nb3 = *reinterpret_cast<const v16bf*>(bp3 + nk);

    acc.a[0][0] = wmma_bf16(a0, b0, acc.a[0][0]);
    acc.a[1][0] = wmma_bf16(a1, b0, acc.a[1][0]);
    acc.a[0][1] = wmma_bf16(a0, b1, acc.a[0][1]);
    acc.a[1][1] = wmma_bf16(a1, b1, acc.a[1][1]);
    acc.a[0][2] = wmma_bf16(a0, b2, acc.a[0][2]);
    acc.a[1][2] = wmma_bf16(a1, b2, acc.a[1][2]);
    acc.a[0][3] = wmma_bf16(a0, b3, acc.a[0][3]);
    acc.a[1][3] = wmma_bf16(a1, b3, acc.a[1][3]);

    a0 = na0; a1 = na1; b0 = nb0; b1 = nb1; b2 = nb2; b3 = nb3;
  }
  return acc;
}

// ---------- kernel 2: fused QKV projection ----------

__global__ void mha_qkv_gemm_kernel(const unsigned short* __restrict__ xbf,
                                    const unsigned short* __restrict__ wT,
                                    const float* __restrict__ bq,
                                    const float* __restrict__ bk,
                                    const float* __restrict__ bv,
                                    unsigned short* __restrict__ q,   // [B,H,S,d]
                                    unsigned short* __restrict__ k,   // [B,H,S,d]
                                    unsigned short* __restrict__ vT)  // [B,H,d,S]
{
  const int lane = threadIdx.x & 31;
  const int gw   = blockIdx.x * (blockDim.x >> 5) + (threadIdx.x >> 5);
  const int which   = gw / (128 * 16);
  const int rem     = gw % (128 * 16);
  const int rowbase = (rem >> 4) * 32;
  const int colbase = (rem & 15) * 64;
  const int half = lane >> 4;
  const int l16  = lane & 15;

  TileAcc acc = gemm_32x64(xbf, wT + (size_t)which * WELEMS,
                           rowbase, colbase, half, l16);

  const float* bias = (which == 0) ? bq : (which == 1) ? bk : bv;
#pragma unroll
  for (int t = 0; t < 4; ++t) {
    const int col = colbase + t * 16 + l16;
    const int hh  = col >> 6, dd = col & 63;
    const float bconst = bias[col];
#pragma unroll
    for (int i = 0; i < 2; ++i) {
#pragma unroll
      for (int r = 0; r < 8; ++r) {
        const int row = rowbase + i * 16 + r + half * 8;
        const int bb = row >> 11, ss = row & 2047;
        const size_t bh = (size_t)(bb * NUM_HEADS + hh);
        const unsigned short val = f2bf(acc.a[i][t][r] + bconst);
        if (which == 2)      vT[(bh * HEAD_SIZE + dd) * SEQ_LEN + ss] = val;
        else if (which == 0) q [(bh * SEQ_LEN + ss) * HEAD_SIZE + dd] = val;
        else                 k [(bh * SEQ_LEN + ss) * HEAD_SIZE + dd] = val;
      }
    }
  }
}

// ---------- kernel 3: fused causal attention ----------
// block = 4 waves sharing one (b,h); waves take 4 consecutive 16-query blocks.
// K/V tiles staged cooperatively in LDS (async global->LDS when available).

__global__ void mha_attention_kernel(const unsigned short* __restrict__ q,
                                     const unsigned short* __restrict__ k,
                                     const unsigned short* __restrict__ vT,
                                     unsigned short* __restrict__ attnout) // [B,S,D]
{
  __shared__ alignas(32) unsigned short kT[32][64];       // [key][d]
  __shared__ alignas(32) unsigned short vS[64][32];       // [d][key]
  __shared__ alignas(16) unsigned short pbuf[4][16][32];  // per-wave P tile

  const int tid  = threadIdx.x;
  const int lane = tid & 31;
  const int w    = tid >> 5;
  const int bh     = blockIdx.x >> 5;            // 0..31
  const int qgroup = blockIdx.x & 31;            // 0..31
  const int qbase  = (qgroup * 4 + w) * 16;
  const int half = lane >> 4;
  const int l16  = lane & 15;

  // Q tile as two A-operands (d = 64 -> K-steps of 32)
  v16bf qa[2];
  {
    const unsigned short* qrow =
        q + ((size_t)bh * SEQ_LEN + qbase + l16) * HEAD_SIZE;
    qa[0] = load_a32(qrow + 0  + half * 16);
    qa[1] = load_a32(qrow + 32 + half * 16);
  }

  float m[8], lsum[8];
  v8f o[4] = {};
#pragma unroll
  for (int r = 0; r < 8; ++r) { m[r] = -1e30f; lsum[r] = 0.f; }

  const int kend_wave  = qbase + 16;
  const int kend_block = (qgroup * 4 + 3) * 16 + 16;  // block-uniform

  for (int kb = 0; kb < kend_block; kb += 32) {
    // ---- cooperative staging: K tile 32x64, V^T tile 64x32 (8KB total) ----
    {
      const unsigned short* kg = k  + ((size_t)bh * SEQ_LEN + kb) * HEAD_SIZE;
      const unsigned short* vg = vT + (size_t)bh * HEAD_SIZE * SEQ_LEN + kb;
#pragma unroll
      for (int c = 0; c < 2; ++c) {
        const int chunk = tid * 2 + c;              // 0..255, 16B each
        const int krow = chunk >> 3, kcol = (chunk & 7) * 8;
        const int vrow = chunk >> 2, vcol = (chunk & 3) * 8;
        stage16(kg + (size_t)krow * HEAD_SIZE + kcol, &kT[krow][kcol]);
        stage16(vg + (size_t)vrow * SEQ_LEN + vcol,  &vS[vrow][vcol]);
      }
    }
    stage_fence();
    __syncthreads();

    if (kb < kend_wave) {   // wave-uniform -> EXEC all 1s for WMMA
      // scores S = Q K^T (two 16x16 N-tiles, K-dim = 64)
      v8f c[2] = {};
#pragma unroll
      for (int t = 0; t < 2; ++t) {
        v16bf b0 = *reinterpret_cast<const v16bf*>(&kT[t * 16 + l16][half * 16]);
        v16bf b1 = *reinterpret_cast<const v16bf*>(&kT[t * 16 + l16][32 + half * 16]);
        c[t] = wmma_bf16(qa[0], b0, c[t]);
        c[t] = wmma_bf16(qa[1], b1, c[t]);
      }

      // reference mask: score*tril/sqrt(d) + (-10000 strict upper)
      float s0[8], s1[8];
#pragma unroll
      for (int r = 0; r < 8; ++r) {
        const int i  = qbase + r + half * 8;
        const int j0 = kb + l16;
        const int j1 = kb + 16 + l16;
        s0[r] = (j0 <= i) ? c[0][r] * 0.125f : -10000.f;
        s1[r] = (j1 <= i) ? c[1][r] * 0.125f : -10000.f;
      }

      // online softmax: butterfly reductions stay inside 16-lane row groups
#pragma unroll
      for (int r = 0; r < 8; ++r) {
        float rv = fmaxf(s0[r], s1[r]);
#pragma unroll
        for (int sft = 1; sft < 16; sft <<= 1) rv = fmaxf(rv, __shfl_xor(rv, sft, 32));
        const float mn = fmaxf(m[r], rv);
        const float ef = __expf(m[r] - mn);
        const float p0 = __expf(s0[r] - mn);
        const float p1 = __expf(s1[r] - mn);
        float rs = p0 + p1;
#pragma unroll
        for (int sft = 1; sft < 16; sft <<= 1) rs += __shfl_xor(rs, sft, 32);
        lsum[r] = lsum[r] * ef + rs;
        m[r] = mn;
#pragma unroll
        for (int t = 0; t < 4; ++t) o[t][r] *= ef;
        pbuf[w][r + half * 8][l16]      = f2bf(p0);  // C-layout -> LDS
        pbuf[w][r + half * 8][l16 + 16] = f2bf(p1);
      }

      // reload P in A-operand layout, then O += P V
      v16bf pa = load_a32(&pbuf[w][l16][half * 16]);
#pragma unroll
      for (int t = 0; t < 4; ++t) {
        v16bf b = *reinterpret_cast<const v16bf*>(&vS[t * 16 + l16][half * 16]);
        o[t] = wmma_bf16(pa, b, o[t]);
      }
    }
    __syncthreads();   // protect LDS tiles before next staging round
  }

  // normalize, merge heads, write bf16 for the output projection
  const int bb = bh >> 4, hh = bh & 15;
#pragma unroll
  for (int t = 0; t < 4; ++t) {
#pragma unroll
    for (int r = 0; r < 8; ++r) {
      const int row = qbase + r + half * 8;
      const float val = o[t][r] / lsum[r];
      attnout[((size_t)(bb * SEQ_LEN + row)) * D_MODEL + hh * HEAD_SIZE + t * 16 + l16]
          = f2bf(val);
    }
  }
}

// ---------- kernel 4: output projection ----------

__global__ void mha_out_gemm_kernel(const unsigned short* __restrict__ attnout,
                                    const unsigned short* __restrict__ wToT,
                                    const float* __restrict__ bo,
                                    float* __restrict__ out) {
  const int lane = threadIdx.x & 31;
  const int gw   = blockIdx.x * (blockDim.x >> 5) + (threadIdx.x >> 5);
  const int rowbase = (gw >> 4) * 32;
  const int colbase = (gw & 15) * 64;
  const int half = lane >> 4;
  const int l16  = lane & 15;

  TileAcc acc = gemm_32x64(attnout, wToT, rowbase, colbase, half, l16);

#pragma unroll
  for (int t = 0; t < 4; ++t) {
    const int col = colbase + t * 16 + l16;
    const float bconst = bo[col];
#pragma unroll
    for (int i = 0; i < 2; ++i) {
#pragma unroll
      for (int r = 0; r < 8; ++r) {
        const int row = rowbase + i * 16 + r + half * 8;
        out[(size_t)row * D_MODEL + col] = acc.a[i][t][r] + bconst;
      }
    }
  }
}

// ---------- launcher ----------

extern "C" void kernel_launch(void* const* d_in, const int* in_sizes, int n_in,
                              void* d_out, int out_size, void* d_ws, size_t ws_size,
                              hipStream_t stream) {
  const float* x  = (const float*)d_in[0];
  const float* wq = (const float*)d_in[1];
  const float* wk = (const float*)d_in[2];
  const float* wv = (const float*)d_in[3];
  const float* wo = (const float*)d_in[4];
  const float* bq = (const float*)d_in[5];
  const float* bk = (const float*)d_in[6];
  const float* bv = (const float*)d_in[7];
  const float* bo = (const float*)d_in[8];
  float* out = (float*)d_out;

  // workspace layout (bf16 stored as ushort), 48 MB total
  char* ws = (char*)d_ws;
  const size_t MB8 = (size_t)8 * 1024 * 1024;
  unsigned short* xbf     = (unsigned short*)(ws + 0 * MB8);  // [4096,1024]
  unsigned short* wT      = (unsigned short*)(ws + 1 * MB8);  // 4 x [1024,1024]^T
  unsigned short* qbuf    = (unsigned short*)(ws + 2 * MB8);  // [B,H,S,d]
  unsigned short* kbuf    = (unsigned short*)(ws + 3 * MB8);  // [B,H,S,d]
  unsigned short* vTbuf   = (unsigned short*)(ws + 4 * MB8);  // [B,H,d,S]
  unsigned short* attnout = (unsigned short*)(ws + 5 * MB8);  // [B,S,D]

  // 1) convert inputs to bf16 (weights transposed for B-operand contiguity)
  {
    const int n = ROWS * D_MODEL;            // 4M threads (covers 1M weight elems)
    mha_convert_kernel<<<(n + 255) / 256, 256, 0, stream>>>(
        x, wq, wk, wv, wo, xbf, wT);
  }
  // 2) fused QKV projection: 3 * 128 * 16 = 6144 wave-tiles (32x64), 4 waves/block
  mha_qkv_gemm_kernel<<<6144 / 4, 128, 0, stream>>>(
      xbf, wT, bq, bk, bv, qbuf, kbuf, vTbuf);
  // 3) fused causal attention: 32 (b,h) x 32 q-groups = 1024 blocks of 4 waves
  mha_attention_kernel<<<1024, 128, 0, stream>>>(qbuf, kbuf, vTbuf, attnout);
  // 4) output projection: 128 * 16 = 2048 wave-tiles (32x64), 4 waves/block
  mha_out_gemm_kernel<<<2048 / 4, 128, 0, stream>>>(
      attnout, wT + (size_t)3 * WELEMS, bo, out);
}
